// SimpleSSMBlock_67800353734997
// MI455X (gfx1250) — matrix operand order repaired and checked by reading, not compiled, not run on hardware
//
#include <hip/hip_runtime.h>
#include <hip/hip_bf16.h>
#include <cstdint>
#include <cstddef>

// Problem dims (fixed by the reference)
#define HIDDEN   1024
#define STATE    16
#define INNER    2048
#define BATCH    4
#define SEQ      2048
#define NTOK     (BATCH * SEQ)      // 8192 tokens
#define TWOINNER (2 * INNER)        // 4096

typedef __attribute__((ext_vector_type(16))) __bf16 v16bf;
typedef __attribute__((ext_vector_type(8)))  float  v8f;

// ---------------------------------------------------------------------------
// float -> bf16 (round to nearest even)
// ---------------------------------------------------------------------------
__device__ __forceinline__ __bf16 f2bf(float f) {
  union { float f; uint32_t u; } v; v.f = f;
  uint32_t r = v.u + 0x7FFFu + ((v.u >> 16) & 1u);
  uint16_t h = (uint16_t)(r >> 16);
  return __builtin_bit_cast(__bf16, h);
}

// ---------------------------------------------------------------------------
// Async global->LDS copy of 16 bytes (CDNA5 async engine, tracked by ASYNCcnt).
// VDST operand = LDS byte address, VADDR = 64-bit global address (GV mode).
// Inline asm per CDNA5 bridge doc (portable across toolchain builtin arities).
// ---------------------------------------------------------------------------
__device__ __forceinline__ void async_copy_b128(uint32_t lds_addr, const void* gsrc) {
  asm volatile("global_load_async_to_lds_b128 %0, %1, off"
               :
               : "v"(lds_addr), "v"((uint64_t)(uintptr_t)gsrc)
               : "memory");
}

__device__ __forceinline__ void wait_asynccnt0() {
  asm volatile("s_wait_asynccnt 0" ::: "memory");
}

// ---------------------------------------------------------------------------
// WMMA fragment loaders (ISA 7.12.2 layouts, wave32)
// A (16x32 MxK, 16-bit): lanes 0-15 -> M=lane, K=[0..7]+[16..23];
//                        lanes 16-31 -> M=lane-16, K=[8..15]+[24..31]
// B (32x16 KxN, 16-bit): lanes 0-15 -> N=lane, K=[0..7]+[16..23];
//                        lanes 16-31 -> N=lane-16, K=[8..15]+[24..31]
// Bt is stored as [N][K] (pre-transposed weights) so both are contiguous.
// ---------------------------------------------------------------------------
__device__ __forceinline__ v16bf load_frag_a(const __bf16* __restrict__ base,
                                             int row0, int ldk, int k0, int lane) {
  const __bf16* p = base + (size_t)(row0 + (lane & 15)) * ldk + k0 + ((lane >> 4) << 3);
  union { v16bf v; uint4 q[2]; } f;
  f.q[0] = *reinterpret_cast<const uint4*>(p);        // K +0..7
  f.q[1] = *reinterpret_cast<const uint4*>(p + 16);   // K +16..23
  return f.v;
}

__device__ __forceinline__ v16bf load_frag_b(const __bf16* __restrict__ base,
                                             int col0, int ldk, int k0, int lane) {
  const __bf16* p = base + (size_t)(col0 + (lane & 15)) * ldk + k0 + ((lane >> 4) << 3);
  union { v16bf v; uint4 q[2]; } f;
  f.q[0] = *reinterpret_cast<const uint4*>(p);        // K +0..7
  f.q[1] = *reinterpret_cast<const uint4*>(p + 16);   // K +16..23
  return f.v;
}

// ---------------------------------------------------------------------------
// Generic bf16 WMMA GEMM:  C[M,N] (f32) = A[M,K](bf16) * Bt[N,K](bf16)^T (+Cinit)
// Block: 256 threads = 8 waves; wave tile 32x32 (2x2 WMMA tiles);
// block tile 128(M) x 64(N). Requires M%128==0, N%32==0, K%32==0.
// ---------------------------------------------------------------------------
__global__ __launch_bounds__(256) void wmma_gemm_bf16(
    const __bf16* __restrict__ A, const __bf16* __restrict__ Bt,
    const float* __restrict__ Cinit, float* __restrict__ C,
    int M, int N, int K) {
  const int lane = threadIdx.x & 31;
  const int wid  = threadIdx.x >> 5;
  const int m0   = blockIdx.y * 128 + (wid >> 1) * 32;
  const int n0   = blockIdx.x * 64  + (wid & 1) * 32;
  if (m0 + 32 > M || n0 + 32 > N) return;

  const int crow = (lane >> 4) << 3;   // 0 or 8 (C/D layout row offset)
  const int ccol = lane & 15;          // C/D column

  v8f c00 = {0.f,0.f,0.f,0.f,0.f,0.f,0.f,0.f};
  v8f c01 = c00, c10 = c00, c11 = c00;
  if (Cinit) {
#pragma unroll
    for (int r = 0; r < 8; ++r) {
      c00[r] = Cinit[(size_t)(m0 + crow + r) * N + n0 + ccol];
      c01[r] = Cinit[(size_t)(m0 + crow + r) * N + n0 + 16 + ccol];
      c10[r] = Cinit[(size_t)(m0 + 16 + crow + r) * N + n0 + ccol];
      c11[r] = Cinit[(size_t)(m0 + 16 + crow + r) * N + n0 + 16 + ccol];
    }
  }

  for (int k = 0; k < K; k += 32) {
    v16bf a0 = load_frag_a(A,  m0,      K, k, lane);
    v16bf a1 = load_frag_a(A,  m0 + 16, K, k, lane);
    v16bf b0 = load_frag_b(Bt, n0,      K, k, lane);
    v16bf b1 = load_frag_b(Bt, n0 + 16, K, k, lane);
    c00 = __builtin_amdgcn_wmma_f32_16x16x32_bf16(false, a0, false, b0, (short)0, c00, false, false);
    c01 = __builtin_amdgcn_wmma_f32_16x16x32_bf16(false, a0, false, b1, (short)0, c01, false, false);
    c10 = __builtin_amdgcn_wmma_f32_16x16x32_bf16(false, a1, false, b0, (short)0, c10, false, false);
    c11 = __builtin_amdgcn_wmma_f32_16x16x32_bf16(false, a1, false, b1, (short)0, c11, false, false);
  }

#pragma unroll
  for (int r = 0; r < 8; ++r) {
    C[(size_t)(m0 + crow + r) * N + n0 + ccol]           = c00[r];
    C[(size_t)(m0 + crow + r) * N + n0 + 16 + ccol]      = c01[r];
    C[(size_t)(m0 + 16 + crow + r) * N + n0 + ccol]      = c10[r];
    C[(size_t)(m0 + 16 + crow + r) * N + n0 + 16 + ccol] = c11[r];
  }
}

// ---------------------------------------------------------------------------
// Weight prep: W[K][N] f32  ->  Wt[N][K] bf16
// ---------------------------------------------------------------------------
__global__ __launch_bounds__(256) void transpose_to_bf16(
    const float* __restrict__ W, __bf16* __restrict__ Wt, int K, int N) {
  int idx = blockIdx.x * 256 + threadIdx.x;
  if (idx >= K * N) return;
  int k = idx / N, n = idx - k * N;
  Wt[(size_t)n * K + k] = f2bf(W[idx]);
}

// ---------------------------------------------------------------------------
// LayerNorm over HIDDEN per token -> bf16 activations for GEMM1
// ---------------------------------------------------------------------------
__global__ __launch_bounds__(256) void layernorm_bf16_kernel(
    const float* __restrict__ x, const float* __restrict__ scale,
    const float* __restrict__ bias, __bf16* __restrict__ xn) {
  __shared__ float s1[256], s2[256];
  const int row = blockIdx.x;
  const float* xr = x + (size_t)row * HIDDEN;
  float sum = 0.f, sq = 0.f;
  for (int i = threadIdx.x; i < HIDDEN; i += 256) {
    float v = xr[i]; sum += v; sq += v * v;
  }
  s1[threadIdx.x] = sum; s2[threadIdx.x] = sq;
  __syncthreads();
  for (int s = 128; s > 0; s >>= 1) {
    if (threadIdx.x < s) { s1[threadIdx.x] += s1[threadIdx.x + s]; s2[threadIdx.x] += s2[threadIdx.x + s]; }
    __syncthreads();
  }
  const float mu  = s1[0] * (1.0f / HIDDEN);
  const float var = s2[0] * (1.0f / HIDDEN) - mu * mu;
  const float inv = rsqrtf(var + 1e-5f);
  for (int i = threadIdx.x; i < HIDDEN; i += 256)
    xn[(size_t)row * HIDDEN + i] = f2bf((xr[i] - mu) * inv * scale[i] + bias[i]);
}

// ---------------------------------------------------------------------------
// Depthwise conv1d (k=3, pad=1, over seq) + SiLU.
// xz is [NTOK, 4096]: cols [0,2048) = conv input, cols [2048,4096) = z (kept).
// ---------------------------------------------------------------------------
__global__ __launch_bounds__(256) void conv_silu_kernel(
    const float* __restrict__ xz, const float* __restrict__ conv_w,
    const float* __restrict__ conv_b, float* __restrict__ xb,
    __bf16* __restrict__ xb_bf) {
  int idx = blockIdx.x * 256 + threadIdx.x;
  if (idx >= NTOK * INNER) return;
  int d   = idx & (INNER - 1);
  int row = idx >> 11;            // / INNER
  int l   = row & (SEQ - 1);
  size_t base = (size_t)row * TWOINNER + d;
  float x0 = xz[base];
  float xm = (l > 0)       ? xz[base - TWOINNER] : 0.f;
  float xp = (l < SEQ - 1) ? xz[base + TWOINNER] : 0.f;
  float v = conv_w[d * 3 + 0] * xm + conv_w[d * 3 + 1] * x0
          + conv_w[d * 3 + 2] * xp + conv_b[d];
  v = v / (1.f + __expf(-v));     // SiLU
  xb[idx]    = v;
  xb_bf[idx] = f2bf(v);
}

// ---------------------------------------------------------------------------
// dt = softplus(dt_pre + b_dt), in place
// ---------------------------------------------------------------------------
__global__ __launch_bounds__(256) void softplus_bias_kernel(
    float* __restrict__ dt, const float* __restrict__ b_dt) {
  int idx = blockIdx.x * 256 + threadIdx.x;
  if (idx >= NTOK * INNER) return;
  float v = dt[idx] + b_dt[idx & (INNER - 1)];
  dt[idx] = (v > 20.f) ? v : log1pf(__expf(v));
}

// ---------------------------------------------------------------------------
// Selective-scan recurrence + fused output gating.
// One lane per channel d; 16 states in registers. B_t/C_t (32 floats/step)
// are streamed into LDS with the CDNA5 async engine: while step t computes
// from buffer t&1, global_load_async_to_lds_b128 prefetches step t+1 into
// the other buffer (ASYNCcnt + barrier synchronization, double buffered).
// Emits y_bf = (y + D*xb)*silu(z).  Grid: (INNER/256, BATCH), block 256.
// ---------------------------------------------------------------------------
__global__ __launch_bounds__(256) void scan_gate_kernel(
    const float* __restrict__ xb, const float* __restrict__ dt,
    const float* __restrict__ bc, const float* __restrict__ A_log,
    const float* __restrict__ xz, const float* __restrict__ Dp,
    __bf16* __restrict__ ybf) {
  const int d = blockIdx.x * 256 + threadIdx.x;
  const int b = blockIdx.y;

  float Ad[STATE];
#pragma unroll
  for (int n = 0; n < STATE; ++n) Ad[n] = -__expf(A_log[(size_t)d * STATE + n]);
  const float Dd = Dp[d];

  float h[STATE];
#pragma unroll
  for (int n = 0; n < STATE; ++n) h[n] = 0.f;

  __shared__ float sBC[2][2 * STATE];
  // LDS byte offset of the buffers (generic LDS pointer low 32 bits = LDS addr)
  const uint32_t lds0 = (uint32_t)(uintptr_t)&sBC[0][0];
  const float* bc_b = bc + (size_t)b * SEQ * (2 * STATE);

  // Prologue: async-stage B/C for t=0 into buffer 0 (8 lanes x 16B = 128B).
  if (threadIdx.x < 8)
    async_copy_b128(lds0 + threadIdx.x * 16, bc_b + threadIdx.x * 4);

  for (int t = 0; t < SEQ; ++t) {
    const size_t row = (size_t)b * SEQ + t;
    wait_asynccnt0();       // issuing wave: staged data landed (no-op for others)
    __syncthreads();        // make buffer t&1 visible to all waves

    // Kick off async stage of step t+1 into the other buffer.
    if (t + 1 < SEQ && threadIdx.x < 8)
      async_copy_b128(lds0 + ((t + 1) & 1) * (2 * STATE * 4) + threadIdx.x * 16,
                      bc_b + (size_t)(t + 1) * (2 * STATE) + threadIdx.x * 4);

    const float* Bv = sBC[t & 1];
    const float* Cv = sBC[t & 1] + STATE;

    const float dv = dt[row * INNER + d];
    const float xv = xb[row * INNER + d];
    const float dx = dv * xv;
    float acc = 0.f;
#pragma unroll
    for (int n = 0; n < STATE; ++n) {
      const float ab = __expf(dv * Ad[n]);
      h[n] = ab * h[n] + dx * Bv[n];
      acc += h[n] * Cv[n];
    }
    const float zz = xz[row * TWOINNER + INNER + d];
    const float g  = zz / (1.f + __expf(-zz));          // silu(z)
    ybf[row * INNER + d] = f2bf((acc + Dd * xv) * g);
  }
}

// ---------------------------------------------------------------------------
// Host orchestration
// ---------------------------------------------------------------------------
extern "C" void kernel_launch(void* const* d_in, const int* in_sizes, int n_in,
                              void* d_out, int out_size, void* d_ws, size_t ws_size,
                              hipStream_t stream) {
  (void)in_sizes; (void)n_in; (void)out_size; (void)ws_size;
  const float* x       = (const float*)d_in[0];
  const float* nscale  = (const float*)d_in[1];
  const float* nbias   = (const float*)d_in[2];
  const float* W_in    = (const float*)d_in[3];   // [1024, 4096]
  const float* conv_w  = (const float*)d_in[4];   // [2048, 3]
  const float* conv_b  = (const float*)d_in[5];
  const float* W_dt    = (const float*)d_in[6];   // [2048, 2048]
  const float* b_dt    = (const float*)d_in[7];
  const float* A_log   = (const float*)d_in[8];   // [2048, 16]
  const float* D_param = (const float*)d_in[9];
  const float* W_B     = (const float*)d_in[10];  // [2048, 16]
  const float* W_C     = (const float*)d_in[11];  // [2048, 16]
  const float* W_out   = (const float*)d_in[12];  // [2048, 1024]
  float* out = (float*)d_out;

  char* ws = (char*)d_ws;
  size_t off = 0;
  auto alloc = [&](size_t bytes) -> char* {
    char* p = ws + off;
    off += (bytes + 255) & ~(size_t)255;
    return p;
  };

  __bf16* Wt_in  = (__bf16*)alloc((size_t)TWOINNER * HIDDEN * sizeof(__bf16)); // [4096][1024]
  __bf16* Wt_dt  = (__bf16*)alloc((size_t)INNER * INNER * sizeof(__bf16));     // [2048][2048]
  __bf16* Wt_bc  = (__bf16*)alloc((size_t)(2 * STATE) * INNER * sizeof(__bf16)); // [32][2048]
  __bf16* Wt_out = (__bf16*)alloc((size_t)HIDDEN * INNER * sizeof(__bf16));    // [1024][2048]
  __bf16* xn_bf  = (__bf16*)alloc((size_t)NTOK * HIDDEN * sizeof(__bf16));
  float*  xz     = (float*) alloc((size_t)NTOK * TWOINNER * sizeof(float));
  float*  xb     = (float*) alloc((size_t)NTOK * INNER * sizeof(float));
  __bf16* xb_bf  = (__bf16*)alloc((size_t)NTOK * INNER * sizeof(__bf16));
  float*  dt     = (float*) alloc((size_t)NTOK * INNER * sizeof(float));
  float*  bc     = (float*) alloc((size_t)NTOK * (2 * STATE) * sizeof(float));
  __bf16* y_bf   = (__bf16*)alloc((size_t)NTOK * INNER * sizeof(__bf16));

  // --- Weight prep (transpose + bf16) ---
  {
    int tot = HIDDEN * TWOINNER;
    transpose_to_bf16<<<(tot + 255) / 256, 256, 0, stream>>>(W_in, Wt_in, HIDDEN, TWOINNER);
  }
  {
    int tot = INNER * INNER;
    transpose_to_bf16<<<(tot + 255) / 256, 256, 0, stream>>>(W_dt, Wt_dt, INNER, INNER);
  }
  {
    int tot = INNER * STATE;
    transpose_to_bf16<<<(tot + 255) / 256, 256, 0, stream>>>(W_B, Wt_bc, INNER, STATE);
    transpose_to_bf16<<<(tot + 255) / 256, 256, 0, stream>>>(W_C, Wt_bc + (size_t)STATE * INNER, INNER, STATE);
  }
  {
    int tot = INNER * HIDDEN;
    transpose_to_bf16<<<(tot + 255) / 256, 256, 0, stream>>>(W_out, Wt_out, INNER, HIDDEN);
  }

  // --- LayerNorm -> bf16 ---
  layernorm_bf16_kernel<<<NTOK, 256, 0, stream>>>(x, nscale, nbias, xn_bf);

  // --- GEMM1: xz = xn @ W_in  [8192 x 4096, K=1024] ---
  wmma_gemm_bf16<<<dim3(TWOINNER / 64, NTOK / 128), 256, 0, stream>>>(
      xn_bf, Wt_in, nullptr, xz, NTOK, TWOINNER, HIDDEN);

  // --- conv(3) + SiLU -> xb (f32 + bf16) ---
  conv_silu_kernel<<<(NTOK * INNER + 255) / 256, 256, 0, stream>>>(
      xz, conv_w, conv_b, xb, xb_bf);

  // --- GEMM2: dt_pre = xb @ W_dt  [8192 x 2048, K=2048], then softplus ---
  wmma_gemm_bf16<<<dim3(INNER / 64, NTOK / 128), 256, 0, stream>>>(
      xb_bf, Wt_dt, nullptr, dt, NTOK, INNER, INNER);
  softplus_bias_kernel<<<(NTOK * INNER + 255) / 256, 256, 0, stream>>>(dt, b_dt);

  // --- GEMM3: [B|C] = xb @ [W_B|W_C]  [8192 x 32, K=2048] ---
  wmma_gemm_bf16<<<dim3(1, NTOK / 128), 256, 0, stream>>>(
      xb_bf, Wt_bc, nullptr, bc, NTOK, 2 * STATE, INNER);

  // --- selective scan + gating -> y_bf (async B/C streaming) ---
  scan_gate_kernel<<<dim3(INNER / 256, BATCH), 256, 0, stream>>>(
      xb, dt, bc, A_log, xz, D_param, y_bf);

  // --- GEMM4: out = x + y @ W_out  [8192 x 1024, K=2048], residual fused via Cinit ---
  wmma_gemm_bf16<<<dim3(HIDDEN / 64, NTOK / 128), 256, 0, stream>>>(
      y_bf, Wt_out, x, out, NTOK, HIDDEN, INNER);
}
